// DeepSeekMoE_3796751090028
// MI455X (gfx1250) — matrix operand order repaired and checked by reading, not compiled, not run on hardware
//
#include <hip/hip_runtime.h>
#include <hip/hip_bf16.h>
#include <math.h>

// ---------------------------------------------------------------------------
// DeepSeek MoE for MI455X (gfx1250): routed bf16 WMMA FFN, fp32 accumulate.
// T=2048 tokens, H=1024, E=8 experts top-2, F=4096, + shared expert (x0.1).
// Double-buffered k-loop: A tiles via global_load_async_to_lds_b128 (ASYNCcnt),
// B (fp32 weights) via register staging + v_cvt to bf16, one barrier/k-step.
// ---------------------------------------------------------------------------

typedef __attribute__((ext_vector_type(16))) __bf16 v16bf;
typedef __attribute__((ext_vector_type(8)))  float  v8f;

constexpr int T   = 2048;     // B*S
constexpr int H   = 1024;
constexpr int F   = 4096;
constexpr int E   = 8;
constexpr int TM  = 128;      // GEMM tile M
constexpr int TN  = 128;      // GEMM tile N
constexpr int TK  = 32;       // GEMM tile K (bf16 WMMA K)
constexpr int CAP = 3 * T + 9 * TM;   // padded row capacity (7296, 57 tiles)
constexpr int LDSS = 40;      // LDS row stride in bf16 elems (80B, 16B-mult)

struct Route { int e0, e1, s0, s1; float w0, w1; };

// ---------------------------------------------------------------------------
__global__ void init_kernel(int* cnt) {
    if (threadIdx.x < 16) cnt[threadIdx.x] = 0;
}

// x fp32 -> bf16, coalesced, 4 elems/thread
__global__ __launch_bounds__(256) void cvt_kernel(const float* __restrict__ x,
                                                  __bf16* __restrict__ xb) {
    int i = (blockIdx.x * 256 + threadIdx.x) * 4;
    float4 f = *reinterpret_cast<const float4*>(x + i);
    xb[i + 0] = (__bf16)f.x;
    xb[i + 1] = (__bf16)f.y;
    xb[i + 2] = (__bf16)f.z;
    xb[i + 3] = (__bf16)f.w;
}

// One wave (32 lanes) per token: 8 gate dots over H, sigmoid, top-2, append.
__global__ __launch_bounds__(256) void gate_kernel(
    const float* __restrict__ x, const float* __restrict__ gw,
    const float* __restrict__ gb, int* __restrict__ cnt,
    int* __restrict__ lists, Route* __restrict__ route) {
    const int wave = threadIdx.x >> 5;
    const int lane = threadIdx.x & 31;
    const int t = blockIdx.x * 8 + wave;
    if (t >= T) return;

    float acc[E];
#pragma unroll
    for (int e = 0; e < E; ++e) acc[e] = 0.0f;
    const float* xr = x + (size_t)t * H;
    for (int h = lane; h < H; h += 32) {
        float xv = xr[h];
#pragma unroll
        for (int e = 0; e < E; ++e) acc[e] += xv * gw[e * H + h];
    }
#pragma unroll
    for (int e = 0; e < E; ++e)
        for (int off = 16; off > 0; off >>= 1)
            acc[e] += __shfl_xor(acc[e], off, 32);

    if (lane == 0) {
        float s[E];
#pragma unroll
        for (int e = 0; e < E; ++e) s[e] = 1.0f / (1.0f + expf(-(acc[e] + gb[e])));
        int b0 = 0;
        for (int e = 1; e < E; ++e) if (s[e] > s[b0]) b0 = e;
        int b1 = -1;
        for (int e = 0; e < E; ++e) {
            if (e == b0) continue;
            if (b1 < 0 || s[e] > s[b1]) b1 = e;
        }
        float sum = s[b0] + s[b1] + 1e-6f;
        float w0 = s[b0] / sum, w1 = s[b1] / sum;
        int slot0 = atomicAdd(&cnt[b0], 1); lists[b0 * T + slot0] = t;
        int slot1 = atomicAdd(&cnt[b1], 1); lists[b1 * T + slot1] = t;
        route[t] = Route{b0, b1, slot0, slot1, w0, w1};
    }
}

// Padded prefix offsets; expert 8 = shared (all T tokens).
__global__ void offsets_kernel(const int* __restrict__ cnt, int* __restrict__ offPad) {
    if (threadIdx.x == 0 && blockIdx.x == 0) {
        int o = 0;
        for (int e = 0; e < E; ++e) {
            offPad[e] = o;
            o += (cnt[e] + TM - 1) & ~(TM - 1);
        }
        offPad[8] = o;
        offPad[9] = o + T;
    }
}

// Global-row -> token id gather table (padding rows map to token 0).
__global__ __launch_bounds__(256) void rowtok_kernel(
    const int* __restrict__ offPad, const int* __restrict__ cnt,
    const int* __restrict__ lists, int* __restrict__ rowTok) {
    int g = blockIdx.x * 256 + threadIdx.x;
    if (g >= CAP) return;
    int total = offPad[9];
    if (g >= total) { rowTok[g] = 0; return; }
    int e = 0;
    while (e < 8 && offPad[e + 1] <= g) ++e;
    int slot = g - offPad[e];
    int tok;
    if (e == 8) tok = slot;
    else tok = (slot < cnt[e]) ? lists[e * T + slot] : 0;
    rowTok[g] = tok;
}

// ---------------------------------------------------------------------------
// Tiled GEMM: D[row, col] = act( A[row,:] (bf16) x W[col,:] (fp32->bf16) + bias[col] )
//   GATHER:   A row index comes through rowTok (up-proj gathers token rows)
//   GELU_ACT: exact GELU epilogue (up-proj)
//   OUT_BF16: store bf16 (a1) vs fp32 (y)
// Double-buffered pipeline, async A copies, one barrier per k-step.
// 8 waves: wave = (wm 0..3) x (wn 0..1); each wave: 2 Mx16 x 4 Nx16 WMMA tiles.
// ---------------------------------------------------------------------------
template <bool GATHER, bool GELU_ACT, bool OUT_BF16>
__global__ __launch_bounds__(256) void ffn_gemm(
    const __bf16* __restrict__ A, int lda,
    const float* __restrict__ Wexp, const float* __restrict__ Wsh, int ldb,
    const float* __restrict__ bExp, const float* __restrict__ bSh,
    const int* __restrict__ offPad, const int* __restrict__ rowTok,
    __bf16* __restrict__ outB, float* __restrict__ outF, int ncols) {
    __shared__ __align__(16) __bf16 As[2][TM][LDSS];
    __shared__ __align__(16) __bf16 Bs[2][TN][LDSS];

    const int rtBase = blockIdx.y * TM;
    const int total = offPad[9];
    if (rtBase >= total) return;
    int e = 0;
    while (e < 8 && offPad[e + 1] <= rtBase) ++e;

    const float* Wb   = (e < 8) ? (Wexp + (size_t)e * (size_t)ncols * (size_t)ldb) : Wsh;
    const float* bias = (e < 8) ? (bExp + (size_t)e * ncols) : bSh;

    const int nBase = blockIdx.x * TN;
    const int tid   = threadIdx.x;
    const int lane  = tid & 31;
    const int wave  = tid >> 5;
    const int wm    = wave >> 1;   // 0..3 -> 32 rows each
    const int wn    = wave & 1;    // 0..1 -> 64 cols each

    v8f acc[2][4];
#pragma unroll
    for (int i = 0; i < 2; ++i)
#pragma unroll
        for (int j = 0; j < 4; ++j) acc[i][j] = (v8f){0, 0, 0, 0, 0, 0, 0, 0};

    // Global->LDS load assignments: 256 threads, 128 rows x 2 halves (32B each).
    const int ar = tid >> 1, ah = tid & 1;
    size_t aRowSrc = GATHER ? (size_t)rowTok[rtBase + ar] : (size_t)(rtBase + ar);
    const __bf16* aSrc = A + aRowSrc * (size_t)lda + ah * 16;     // + kb
    const float*  bSrc = Wb + (size_t)(nBase + ar) * (size_t)ldb + ah * 16;  // + kb

    // --- async copy of this thread's 32B slice of the A tile into As[b] ---
    auto issueA = [&](int kb, int b) {
        unsigned lds = (unsigned)(unsigned long long)(const void*)&As[b][ar][ah * 16];
        unsigned long long g = (unsigned long long)(const void*)(aSrc + kb);
        asm volatile(
            "global_load_async_to_lds_b128 %0, %1, off\n\t"
            "global_load_async_to_lds_b128 %0, %1, off offset:16"
            :: "v"(lds), "v"(g) : "memory");
    };
    // --- stage 16 fp32 weight values into registers ---
    float4 bReg0, bReg1, bReg2, bReg3;
    auto loadB = [&](int kb) {
        const float4* src = reinterpret_cast<const float4*>(bSrc + kb);
        bReg0 = src[0]; bReg1 = src[1]; bReg2 = src[2]; bReg3 = src[3];
    };
    // --- convert staged weights to bf16 and store to Bs[b] ---
    auto cvtStoreB = [&](int b) {
        union { __bf16 h[16]; uint4 u[2]; } tmp;
        tmp.h[0]  = (__bf16)bReg0.x; tmp.h[1]  = (__bf16)bReg0.y;
        tmp.h[2]  = (__bf16)bReg0.z; tmp.h[3]  = (__bf16)bReg0.w;
        tmp.h[4]  = (__bf16)bReg1.x; tmp.h[5]  = (__bf16)bReg1.y;
        tmp.h[6]  = (__bf16)bReg1.z; tmp.h[7]  = (__bf16)bReg1.w;
        tmp.h[8]  = (__bf16)bReg2.x; tmp.h[9]  = (__bf16)bReg2.y;
        tmp.h[10] = (__bf16)bReg2.z; tmp.h[11] = (__bf16)bReg2.w;
        tmp.h[12] = (__bf16)bReg3.x; tmp.h[13] = (__bf16)bReg3.y;
        tmp.h[14] = (__bf16)bReg3.z; tmp.h[15] = (__bf16)bReg3.w;
        uint4* dst = reinterpret_cast<uint4*>(&Bs[b][ar][ah * 16]);
        dst[0] = tmp.u[0]; dst[1] = tmp.u[1];
    };

    // Fragment layouts per ISA (wave32):
    //  A 16x32: lanes 0-15 hold K 0-7 & 16-23 (M=lane); lanes 16-31 K 8-15 & 24-31.
    //  B 32x16: lanes 0-15 hold K 0-7 & 16-23 (N=lane&15); lanes 16-31 K 8-15 & 24-31.
    const int am   = lane & 15;
    const int aOff = (lane < 16) ? 0 : 8;
    const int bOff = (lane < 16) ? 0 : 8;
    auto compute = [&](int b) {
        v16bf afrag[2], bfrag[4];
#pragma unroll
        for (int mt = 0; mt < 2; ++mt) {
            int row = wm * 32 + mt * 16 + am;
            union { uint4 u[2]; v16bf v; } fa;
            fa.u[0] = *reinterpret_cast<const uint4*>(&As[b][row][aOff]);
            fa.u[1] = *reinterpret_cast<const uint4*>(&As[b][row][aOff + 16]);
            afrag[mt] = fa.v;
        }
#pragma unroll
        for (int nt = 0; nt < 4; ++nt) {
            int col = wn * 64 + nt * 16 + am;
            union { uint4 u[2]; v16bf v; } fb;
            fb.u[0] = *reinterpret_cast<const uint4*>(&Bs[b][col][bOff]);
            fb.u[1] = *reinterpret_cast<const uint4*>(&Bs[b][col][bOff + 16]);
            bfrag[nt] = fb.v;
        }
#pragma unroll
        for (int mt = 0; mt < 2; ++mt)
#pragma unroll
            for (int nt = 0; nt < 4; ++nt)
                acc[mt][nt] = __builtin_amdgcn_wmma_f32_16x16x32_bf16(
                    false, afrag[mt], false, bfrag[nt], (short)0, acc[mt][nt],
                    false, false);
    };

    const int Klen = ldb;

    // Prologue: fill buffer 0 with tile kb=0.
    issueA(0, 0);
    loadB(0);
    cvtStoreB(0);
    asm volatile("s_wait_asynccnt 0x0" ::: "memory");
    __syncthreads();

    int buf = 0;
    for (int kb = 0; kb < Klen; kb += TK) {
        const int nxt = buf ^ 1;
        const bool hasNext = (kb + TK < Klen);
        if (hasNext) {
            issueA(kb + TK, nxt);             // async -> As[nxt] (ASYNCcnt)
            loadB(kb + TK);                    // fp32 weights -> regs (LOADcnt)
            if (kb + 2 * TK < Klen)
                __builtin_prefetch(bSrc + kb + 2 * TK, 0, 0);
        }
        compute(buf);                          // ds_load frags + 8x WMMA
        if (hasNext) {
            cvtStoreB(nxt);                    // v_cvt + ds_store -> Bs[nxt]
            asm volatile("s_wait_asynccnt 0x0" ::: "memory");
        }
        __syncthreads();                       // one barrier per k-step
        buf = nxt;
    }

    // --- Epilogue: D layout: lane n = lane&15, VGPR v -> M = v + (lane<16?0:8)
    const int mHi = (lane < 16) ? 0 : 8;
#pragma unroll
    for (int mt = 0; mt < 2; ++mt) {
#pragma unroll
        for (int nt = 0; nt < 4; ++nt) {
            int col = nBase + wn * 64 + nt * 16 + (lane & 15);
            float bv = bias[col];
#pragma unroll
            for (int v = 0; v < 8; ++v) {
                int row = rtBase + wm * 32 + mt * 16 + v + mHi;
                float val = acc[mt][nt][v] + bv;
                if (GELU_ACT)
                    val = 0.5f * val * (1.0f + erff(val * 0.70710678118654752f));
                if (OUT_BF16)
                    outB[(size_t)row * ncols + col] = (__bf16)val;
                else
                    outF[(size_t)row * ncols + col] = val;
            }
        }
    }
}

// out[t] = w0*y[row(e0,s0)] + w1*y[row(e1,s1)] + 0.1*y[shared row t]
__global__ __launch_bounds__(256) void combine_kernel(
    const float* __restrict__ y, const Route* __restrict__ route,
    const int* __restrict__ offPad, float* __restrict__ out) {
    const int t = blockIdx.x;
    const int h = threadIdx.x * 4;
    Route r = route[t];
    const float4 a = *reinterpret_cast<const float4*>(y + (size_t)(offPad[r.e0] + r.s0) * H + h);
    const float4 b = *reinterpret_cast<const float4*>(y + (size_t)(offPad[r.e1] + r.s1) * H + h);
    const float4 c = *reinterpret_cast<const float4*>(y + (size_t)(offPad[8] + t) * H + h);
    float4 o;
    o.x = r.w0 * a.x + r.w1 * b.x + 0.1f * c.x;
    o.y = r.w0 * a.y + r.w1 * b.y + 0.1f * c.y;
    o.z = r.w0 * a.z + r.w1 * b.z + 0.1f * c.z;
    o.w = r.w0 * a.w + r.w1 * b.w + 0.1f * c.w;
    *reinterpret_cast<float4*>(out + (size_t)t * H + h) = o;
}

// ---------------------------------------------------------------------------
extern "C" void kernel_launch(void* const* d_in, const int* in_sizes, int n_in,
                              void* d_out, int out_size, void* d_ws, size_t ws_size,
                              hipStream_t stream) {
    const float* x     = (const float*)d_in[0];
    const float* gw    = (const float*)d_in[1];
    const float* gb    = (const float*)d_in[2];
    const float* Wu    = (const float*)d_in[3];
    const float* bu    = (const float*)d_in[4];
    const float* Wd    = (const float*)d_in[5];
    const float* bd    = (const float*)d_in[6];
    const float* sWu   = (const float*)d_in[7];
    const float* sbu   = (const float*)d_in[8];
    const float* sWd   = (const float*)d_in[9];
    const float* sbd   = (const float*)d_in[10];
    float* out = (float*)d_out;

    // Workspace carve (256B aligned slabs)
    char* base = (char*)d_ws;
    size_t off = 0;
    auto carve = [&](size_t bytes) -> char* {
        off = (off + 255) & ~(size_t)255;
        char* p = base + off;
        off += bytes;
        return p;
    };
    __bf16* xb     = (__bf16*)carve((size_t)T * H * sizeof(__bf16));
    __bf16* a1     = (__bf16*)carve((size_t)CAP * F * sizeof(__bf16));
    float*  yb     = (float*)carve((size_t)CAP * H * sizeof(float));
    int*    cnt    = (int*)carve(16 * sizeof(int));
    int*    offPad = (int*)carve(16 * sizeof(int));
    int*    lists  = (int*)carve((size_t)E * T * sizeof(int));
    int*    rowTok = (int*)carve((size_t)CAP * sizeof(int));
    Route*  route  = (Route*)carve((size_t)T * sizeof(Route));
    (void)ws_size; (void)in_sizes; (void)n_in; (void)out_size;

    init_kernel<<<1, 64, 0, stream>>>(cnt);
    cvt_kernel<<<(T * H) / (256 * 4), 256, 0, stream>>>(x, xb);
    gate_kernel<<<T / 8, 256, 0, stream>>>(x, gw, gb, cnt, lists, route);
    offsets_kernel<<<1, 1, 0, stream>>>(cnt, offPad);
    rowtok_kernel<<<(CAP + 255) / 256, 256, 0, stream>>>(offPad, cnt, lists, rowTok);

    // Up-projection + GELU: A = gathered bf16 x rows, W = Wu[e] (F x H), out a1 bf16
    ffn_gemm<true, true, true><<<dim3(F / TN, CAP / TM), 256, 0, stream>>>(
        xb, H, Wu, sWu, H, bu, sbu, offPad, rowTok, a1, nullptr, F);

    // Down-projection: A = a1 rows (dense), W = Wd[e] (H x F), out y fp32 (+bd)
    ffn_gemm<false, false, false><<<dim3(H / TN, CAP / TM), 256, 0, stream>>>(
        a1, F, Wd, sWd, F, bd, sbd, offPad, rowTok, nullptr, yb, H);

    combine_kernel<<<T, 256, 0, stream>>>(yb, route, offPad, out);
}
